// ExperimentalLayer7_1563368096370
// MI455X (gfx1250) — matrix operand order repaired and checked by reading, not compiled, not run on hardware
//
#include <hip/hip_runtime.h>
#include <stdint.h>

typedef __attribute__((ext_vector_type(16))) __bf16 bf16x16;
typedef __attribute__((ext_vector_type(8)))  __bf16 bf16x8;
typedef __attribute__((ext_vector_type(8)))  float  f32x8;

// Explicit global address space so fragment loads lower to global_load_b128
// (LOADcnt only) instead of flat_load_b128 (LOADcnt + DScnt coupling).
#define GLOBAL_AS __attribute__((address_space(1)))
typedef const __bf16  GLOBAL_AS* gbf16cp;
typedef const bf16x8  GLOBAL_AS* gbf16x8cp;
typedef const bf16x16 GLOBAL_AS* gbf16x16cp;

#define WMMA_BF16(a, b, c) \
  __builtin_amdgcn_wmma_f32_16x16x32_bf16(false, (a), false, (b), (short)0, (c), false, false)

static constexpr int kB  = 4;
static constexpr int kS  = 2048;
static constexpr int kD  = 1024;

// ---------------------------------------------------------------------------
// fp32 -> bf16 conversion (RNE via compiler fptrunc)
// ---------------------------------------------------------------------------
__global__ __launch_bounds__(256) void cvt_bf16(const float* __restrict__ in,
                                                __bf16* __restrict__ out, int n) {
  int i = blockIdx.x * 256 + threadIdx.x;
  if (i < n) out[i] = (__bf16)in[i];
}

// ---------------------------------------------------------------------------
// 64x64 wave-tile GEMM core: C[64x64] += A[64xK] * B[64xK]^T
// A rows: lda-strided, contiguous in k.  B rows: ldb-strided, contiguous in k
// (B row n supplies WMMA B-matrix column n).
// WMMA bf16 16x16x32 fragment layout (wave32):
//   A: lane<16 holds row (lane&15), k in {0..7,16..23}; lane>=16: {8..15,24..31}
//   B: lane<16 holds col (lane&15), k in {0..7,16..23}; lane>=16: {8..15,24..31}
//   C: lane<16 col lane, rows 0..7 in v0..7; lane>=16 rows 8..15
// ---------------------------------------------------------------------------
__device__ inline void wave_gemm_64x64(const __bf16* __restrict__ Abase, size_t lda,
                                       const __bf16* __restrict__ Bbase, size_t ldb,
                                       int kiters, int lane, f32x8 acc[4][4]) {
  const int lhalf = lane >> 4;     // 0 or 1
  const int lrow  = lane & 15;
  gbf16cp pa[4];
  gbf16cp pb[4];
#pragma unroll
  for (int mi = 0; mi < 4; ++mi)
    pa[mi] = (gbf16cp)(Abase + (size_t)(mi * 16 + lrow) * lda + lhalf * 8);
#pragma unroll
  for (int ni = 0; ni < 4; ++ni)
    pb[ni] = (gbf16cp)(Bbase + (size_t)(ni * 16 + lrow) * ldb + lhalf * 8);

  for (int kk = 0; kk < kiters; ++kk) {
    bf16x16 av[4], bv[4];
#pragma unroll
    for (int mi = 0; mi < 4; ++mi) {
      bf16x8 lo = *(gbf16x8cp)(pa[mi]);        // k-chunk 0..7 (or 8..15)
      bf16x8 hi = *(gbf16x8cp)(pa[mi] + 16);   // k-chunk 16..23 (or 24..31)
#pragma unroll
      for (int e = 0; e < 8; ++e) { av[mi][e] = lo[e]; av[mi][e + 8] = hi[e]; }
      pa[mi] += 32;
    }
#pragma unroll
    for (int ni = 0; ni < 4; ++ni) {
      bf16x8 blo = *(gbf16x8cp)(pb[ni]);
      bf16x8 bhi = *(gbf16x8cp)(pb[ni] + 16);
#pragma unroll
      for (int e = 0; e < 8; ++e) { bv[ni][e] = blo[e]; bv[ni][e + 8] = bhi[e]; }
      pb[ni] += 32;
    }
#pragma unroll
    for (int mi = 0; mi < 4; ++mi)
#pragma unroll
      for (int ni = 0; ni < 4; ++ni)
        acc[mi][ni] = WMMA_BF16(av[mi], bv[ni], acc[mi][ni]);
  }
}

// ---------------------------------------------------------------------------
// Kernel 1: kv = x @ W^T.  M = B*S = 8192, N = 2*D = 2048, K = D = 1024.
// j <  D : K  stored row-major  Kb[b][s][j]
// j >= D : V  stored transposed Vt[b][j-D][s]   (free transpose from D-tile)
// Block = 128 threads = 4 waves in 2x2 -> 128x128 block tile.
// ---------------------------------------------------------------------------
__global__ __launch_bounds__(128) void kv_proj(const __bf16* __restrict__ xb,
                                               const __bf16* __restrict__ Wb,
                                               __bf16* __restrict__ Kb,
                                               __bf16* __restrict__ Vt) {
  const int lane = threadIdx.x & 31;
  const int wave = threadIdx.x >> 5;
  const int m0 = blockIdx.x * 128 + (wave & 1) * 64;
  const int n0 = blockIdx.y * 128 + (wave >> 1) * 64;

  f32x8 acc[4][4] = {};
  wave_gemm_64x64(xb + (size_t)m0 * kD, kD, Wb + (size_t)n0 * kD, kD,
                  kD / 32, lane, acc);

  const int lhalf = lane >> 4, lcol = lane & 15;
#pragma unroll
  for (int mi = 0; mi < 4; ++mi)
#pragma unroll
    for (int ni = 0; ni < 4; ++ni) {
      const int ncol = n0 + ni * 16 + lcol;
#pragma unroll
      for (int r = 0; r < 8; ++r) {
        const int row = m0 + mi * 16 + lhalf * 8 + r;   // global b*S+s
        const float v = acc[mi][ni][r];
        if (ncol < kD) {
          Kb[(size_t)row * kD + ncol] = (__bf16)v;
        } else {
          const int b = row >> 11, s = row & (kS - 1), dd = ncol - kD;
          Vt[((size_t)b * kD + dd) * kS + s] = (__bf16)v;
        }
      }
    }
}

// ---------------------------------------------------------------------------
// Kernel 2: scores S[b][q][k] = sum_d x[b][q][d] * K[b][k][d]  (causal tiles only)
// ---------------------------------------------------------------------------
__global__ __launch_bounds__(128) void attn_scores(const __bf16* __restrict__ xb,
                                                   const __bf16* __restrict__ Kb,
                                                   float* __restrict__ Sc) {
  const int lane = threadIdx.x & 31;
  const int wave = threadIdx.x >> 5;
  const int b  = blockIdx.z;
  const int q0 = blockIdx.x * 128 + (wave & 1) * 64;
  const int k0 = blockIdx.y * 128 + (wave >> 1) * 64;
  if (k0 > q0) return;   // wave-uniform: tile fully above diagonal

  f32x8 acc[4][4] = {};
  wave_gemm_64x64(xb + ((size_t)b * kS + q0) * kD, kD,
                  Kb + ((size_t)b * kS + k0) * kD, kD,
                  kD / 32, lane, acc);

  float* Sb = Sc + (size_t)b * kS * kS;
  const int lhalf = lane >> 4, lcol = lane & 15;
#pragma unroll
  for (int mi = 0; mi < 4; ++mi)
#pragma unroll
    for (int ni = 0; ni < 4; ++ni) {
      const int k = k0 + ni * 16 + lcol;
#pragma unroll
      for (int r = 0; r < 8; ++r) {
        const int q = q0 + mi * 16 + lhalf * 8 + r;
        Sb[(size_t)q * kS + k] = acc[mi][ni][r];   // k>q never read downstream
      }
    }
}

// ---------------------------------------------------------------------------
// Kernel 3: row softmax over k in [0, q]; write bf16 P, zeros for k > q.
// One 256-thread block per (b, q) row.
// ---------------------------------------------------------------------------
__global__ __launch_bounds__(256) void softmax_rows(const float* __restrict__ Sc,
                                                    __bf16* __restrict__ P) {
  __shared__ float red[256];
  const int row = blockIdx.x;               // b*S + q
  const int q   = row & (kS - 1);
  const int tid = threadIdx.x;
  const int len = q + 1;
  const float* s = Sc + (size_t)row * kS;

  float m = -3.4e38f;
  for (int i = tid; i < len; i += 256) m = fmaxf(m, s[i]);
  red[tid] = m; __syncthreads();
  for (int w = 128; w > 0; w >>= 1) {
    if (tid < w) red[tid] = fmaxf(red[tid], red[tid + w]);
    __syncthreads();
  }
  m = red[0]; __syncthreads();

  float sum = 0.f;
  for (int i = tid; i < len; i += 256) sum += __expf(s[i] - m);
  red[tid] = sum; __syncthreads();
  for (int w = 128; w > 0; w >>= 1) {
    if (tid < w) red[tid] += red[tid + w];
    __syncthreads();
  }
  const float inv = 1.0f / red[0];

  __bf16* p = P + (size_t)row * kS;
  for (int i = tid; i < kS; i += 256) {
    float v = (i < len) ? __expf(s[i] - m) * inv : 0.0f;
    p[i] = (__bf16)v;
  }
}

// ---------------------------------------------------------------------------
// Kernel 4: out[b][q][n] = x[b][q][n] + sum_k P[b][q][k] * Vt[b][n][k]
// k-loop causally truncated to q0+64 (exact: P has explicit zeros past q).
// ---------------------------------------------------------------------------
__global__ __launch_bounds__(128) void attn_out(const __bf16* __restrict__ P,
                                                const __bf16* __restrict__ Vt,
                                                const float* __restrict__ x,
                                                float* __restrict__ out) {
  const int lane = threadIdx.x & 31;
  const int wave = threadIdx.x >> 5;
  const int b  = blockIdx.z;
  const int q0 = blockIdx.x * 128 + (wave & 1) * 64;
  const int n0 = blockIdx.y * 128 + (wave >> 1) * 64;

  f32x8 acc[4][4] = {};
  const int kiters = (q0 + 64) / 32;
  wave_gemm_64x64(P  + ((size_t)b * kS + q0) * kS, kS,
                  Vt + ((size_t)b * kD + n0) * kS, kS,
                  kiters, lane, acc);

  const int lhalf = lane >> 4, lcol = lane & 15;
#pragma unroll
  for (int mi = 0; mi < 4; ++mi)
#pragma unroll
    for (int ni = 0; ni < 4; ++ni) {
      const int n = n0 + ni * 16 + lcol;
#pragma unroll
      for (int r = 0; r < 8; ++r) {
        const int q = q0 + mi * 16 + lhalf * 8 + r;
        const size_t idx = ((size_t)b * kS + q) * kD + n;
        out[idx] = x[idx] + acc[mi][ni][r];
      }
    }
}

// ---------------------------------------------------------------------------
extern "C" void kernel_launch(void* const* d_in, const int* in_sizes, int n_in,
                              void* d_out, int out_size, void* d_ws, size_t ws_size,
                              hipStream_t stream) {
  (void)in_sizes; (void)n_in; (void)out_size; (void)ws_size;
  const float* x = (const float*)d_in[0];   // (B, S, D) fp32
  const float* W = (const float*)d_in[1];   // (2D, D)   fp32
  float* out = (float*)d_out;               // (B, S, D) fp32

  char* ws = (char*)d_ws;
  size_t off = 0;
  __bf16* xb = (__bf16*)(ws + off); off += (size_t)kB * kS * kD * 2;   // 16 MB
  __bf16* Wb = (__bf16*)(ws + off); off += (size_t)2 * kD * kD * 2;    //  4 MB
  __bf16* Kb = (__bf16*)(ws + off); off += (size_t)kB * kS * kD * 2;   // 16 MB
  __bf16* Vt = (__bf16*)(ws + off); off += (size_t)kB * kS * kD * 2;   // 16 MB
  float*  Sc = (float*) (ws + off); off += (size_t)kB * kS * kS * 4;   // 64 MB
  __bf16* P  = (__bf16*)(ws + off); off += (size_t)kB * kS * kS * 2;   // 32 MB

  const int nx = kB * kS * kD;   // 8388608
  const int nw = 2 * kD * kD;    // 2097152
  cvt_bf16<<<(nx + 255) / 256, 256, 0, stream>>>(x, xb, nx);
  cvt_bf16<<<(nw + 255) / 256, 256, 0, stream>>>(W, Wb, nw);

  kv_proj<<<dim3((kB * kS) / 128, (2 * kD) / 128), 128, 0, stream>>>(xb, Wb, Kb, Vt);
  attn_scores<<<dim3(kS / 128, kS / 128, kB), 128, 0, stream>>>(xb, Kb, Sc);
  softmax_rows<<<kB * kS, 256, 0, stream>>>(Sc, P);
  attn_out<<<dim3(kS / 128, kD / 128, kB), 128, 0, stream>>>(P, Vt, x, out);
}
